// RelativePositionSDPA_90829968376477
// MI455X (gfx1250) — matrix operand order repaired and checked by reading, not compiled, not run on hardware
//
#include <hip/hip_runtime.h>

// ---------------------------------------------------------------------------
// Transformer-XL relative-position SDPA, fused flash-attention style, for
// MI455X (gfx1250, wave32, WMMA).  N=2 H=8 S=1024 DH=64 D=512.
//
// Roofline: ~10.7 GFLOP vs ~22MB unique bytes (~1us at 23.3 TB/s; resident in
// the 192MB L2).  Strategy: never materialize SxS / Sx(2S-1) scores (online
// softmax), run QK^T / rel-pos / PV on v_wmma_f32_16x16x32_bf16 (fp32 accum),
// and the small precision-critical r-projection on v_wmma_f32_16x16x4_f32.
// The XL "shift" is folded in per 16x16 tile: bd[ii,jj]=qv[ii].r[p0+jj-ii]
// touches a 31-row window of r -> one 16x32 WMMA product + 2 lane-permutes
// per VGPR to extract the diagonal band.  bf16 packing uses native __bf16
// conversions so the backend emits hardware cvt ops instead of integer
// emulation (cuts hot-loop VALU).
// ---------------------------------------------------------------------------

#define SEQ 1024
#define DHD 64
#define DMODEL 512
#define NPOS 2047  // 2S-1

typedef __attribute__((ext_vector_type(16))) __bf16 v16bf;
typedef __attribute__((ext_vector_type(8)))  float  v8f;
typedef __attribute__((ext_vector_type(2)))  float  v2f;

union FragB { unsigned int u[8]; v16bf v; };
union FragC { float f[8]; v8f v; };

#if defined(__HIP_DEVICE_COMPILE__) && __has_builtin(__builtin_amdgcn_wmma_f32_16x16x4_f32)
#define HAVE_WMMA_F32 1
#else
#define HAVE_WMMA_F32 0
#endif

__device__ __forceinline__ v8f wmma_bf16(v16bf a, v16bf b, v8f c) {
#if defined(__HIP_DEVICE_COMPILE__)
  return __builtin_amdgcn_wmma_f32_16x16x32_bf16(false, a, false, b, (short)0, c,
                                                 false, false);
#else
  return c;
#endif
}

__device__ __forceinline__ void wavebar() {
#if defined(__HIP_DEVICE_COMPILE__)
  __builtin_amdgcn_wave_barrier();
#endif
}

// native f32 -> bf16 (RNE); backend selects hardware cvt/pack ops on gfx1250
__device__ __forceinline__ unsigned int pack_bf2(float a, float b) {
  union { __bf16 b[2]; unsigned int u; } t;
  t.b[0] = (__bf16)a;
  t.b[1] = (__bf16)b;
  return t.u;
}
__device__ __forceinline__ unsigned short f2bf(float a) {
  union { __bf16 b; unsigned short u; } t;
  t.b = (__bf16)a;
  return t.u;
}

// ---------------- kernel 0: sinusoidal encodings (2047 x 512 fp32) ---------
__global__ void __launch_bounds__(256) enc_kernel(float* __restrict__ enc) {
  int idx = blockIdx.x * 256 + threadIdx.x;
  if (idx >= NPOS * DMODEL) return;
  int p = idx >> 9;
  int c = idx & 511;
  float pos  = (float)(SEQ - 1 - p);
  float t    = (float)(c >> 1);
  float invf = expf(t * (-9.210340371976184f / 256.0f));  // 10000^(-2t/512)
  float ang  = pos * invf;
  enc[idx] = (c & 1) ? cosf(ang) : sinf(ang);
}

// ---------------- kernel 1: r = enc @ W^T  (fp32 WMMA), store bf16 ---------
// output layout Rb[h][p][dh], h = o>>6, dh = o&63
__global__ void __launch_bounds__(256) rproj_kernel(const float* __restrict__ enc,
                                                    const float* __restrict__ W,
                                                    unsigned short* __restrict__ Rb) {
  const int lane = threadIdx.x & 31;
  const int wave = threadIdx.x >> 5;
  const int tile = blockIdx.x * 8 + wave;      // 128 p-tiles x 32 o-tiles
  const int P0 = (tile >> 5) * 16, O0 = (tile & 31) * 16;
  const int lh = lane >> 4, lr = lane & 15;
  float res[8];
#if HAVE_WMMA_F32
  {
    int prow = P0 + lr; if (prow > NPOS - 1) prow = NPOS - 1;
    const float* ea = enc + prow * DMODEL + lh * 2;        // A: 16x4 f32
    const float* wb = W + (O0 + lr) * DMODEL + lh * 2;     // B: 4x16 f32
    FragC c;
#pragma unroll
    for (int i = 0; i < 8; i++) c.f[i] = 0.f;
    for (int k = 0; k < DMODEL; k += 4) {
      v2f a = *(const v2f*)(ea + k);
      v2f b = *(const v2f*)(wb + k);
      c.v = __builtin_amdgcn_wmma_f32_16x16x4_f32(false, a, false, b, (short)0,
                                                  c.v, false, false);
    }
#pragma unroll
    for (int v = 0; v < 8; v++) res[v] = c.f[v];
  }
#else
#pragma unroll
  for (int v = 0; v < 8; v++) {
    int p = P0 + v + lh * 8; if (p > NPOS - 1) p = NPOS - 1;
    int o = O0 + lr;
    float s = 0.f;
    for (int k = 0; k < DMODEL; k++) s += enc[p * DMODEL + k] * W[o * DMODEL + k];
    res[v] = s;
  }
#endif
#pragma unroll
  for (int v = 0; v < 8; v++) {
    int p = P0 + v + lh * 8;
    if (p < NPOS) {
      int o = O0 + lr;
      Rb[(((o >> 6) * NPOS) + p) * DHD + (o & 63)] = f2bf(res[v]);
    }
  }
}

// ---------------- kernel 2: fused causal rel-pos flash attention -----------
__global__ void __launch_bounds__(256) attn_kernel(
    const float* __restrict__ seqs, const float* __restrict__ keys,
    const float* __restrict__ values, const float* __restrict__ ubias,
    const float* __restrict__ vbias, const unsigned short* __restrict__ Rb,
    float* __restrict__ out) {
  __shared__ unsigned int sP[8][256];  // per-wave 16x32 bf16 P staging (1KB/wave)

  const int lane = threadIdx.x & 31;
  const int wave = threadIdx.x >> 5;
  const int wid  = blockIdx.x * 8 + wave;   // 1024 wave-tiles
  const int head = wid >> 6;                // n*8+h
  const int I0   = (wid & 63) << 4;         // query tile base
  const int lh   = lane >> 4;               // lane half
  const int lr   = lane & 15;
  const int aoff = lh * 8;                  // D-layout row offset
  const int boff = lh * 16;                 // B-layout k offset
  const int h    = head & 7;

  const float* Qp = seqs   + (size_t)head * SEQ * DHD;
  const float* Kp = keys   + (size_t)head * SEQ * DHD;
  const float* Vp = values + (size_t)head * SEQ * DHD;
  const unsigned short* Rh = Rb + (size_t)h * NPOS * DHD;

  // Q fragments (A layout), content- and position-biased, held for whole pass
  FragB Qu[2], Qv[2];
  {
    const float* q  = Qp + (I0 + lr) * DHD;
    const float* ub = ubias + h * DHD;
    const float* vb = vbias + h * DHD;
#pragma unroll
    for (int ks = 0; ks < 2; ks++) {
#pragma unroll
      for (int g = 0; g < 2; g++) {
        int d0 = ks * 32 + boff + g * 8;
        float4 qa = *(const float4*)(q + d0);
        float4 qb = *(const float4*)(q + d0 + 4);
        float4 ua = *(const float4*)(ub + d0);
        float4 ub4 = *(const float4*)(ub + d0 + 4);
        float4 va = *(const float4*)(vb + d0);
        float4 vb4 = *(const float4*)(vb + d0 + 4);
        Qu[ks].u[g * 4 + 0] = pack_bf2(qa.x + ua.x, qa.y + ua.y);
        Qu[ks].u[g * 4 + 1] = pack_bf2(qa.z + ua.z, qa.w + ua.w);
        Qu[ks].u[g * 4 + 2] = pack_bf2(qb.x + ub4.x, qb.y + ub4.y);
        Qu[ks].u[g * 4 + 3] = pack_bf2(qb.z + ub4.z, qb.w + ub4.w);
        Qv[ks].u[g * 4 + 0] = pack_bf2(qa.x + va.x, qa.y + va.y);
        Qv[ks].u[g * 4 + 1] = pack_bf2(qa.z + va.z, qa.w + va.w);
        Qv[ks].u[g * 4 + 2] = pack_bf2(qb.x + vb4.x, qb.y + vb4.y);
        Qv[ks].u[g * 4 + 3] = pack_bf2(qb.z + vb4.z, qb.w + vb4.w);
      }
    }
  }

  // ones B-fragment: column 0 all-ones -> softmax denominator via WMMA
  FragB ones;
#pragma unroll
  for (int d = 0; d < 8; d++) ones.u[d] = (lr == 0) ? 0x3f803f80u : 0u;

  FragC acc[4], lsum;
#pragma unroll
  for (int t = 0; t < 4; t++)
#pragma unroll
    for (int v = 0; v < 8; v++) acc[t].f[v] = 0.f;
#pragma unroll
  for (int v = 0; v < 8; v++) lsum.f[v] = 0.f;
  float mrow[8];
#pragma unroll
  for (int v = 0; v < 8; v++) mrow[v] = -1e30f;

  const int nblk = ((I0 + 15) >> 5) + 1;      // causal 32-key blocks
  for (int blk = 0; blk < nblk; blk++) {
    const int Jb = blk * 32;
    FragC s[2];
#pragma unroll
    for (int kt = 0; kt < 2; kt++) {
      const int Jt = Jb + kt * 16;
      // ---- K tile -> B fragments (this lane = key column Jt+lr) ----
      FragB KB[2];
      {
        const float* kp = Kp + (Jt + lr) * DHD;
#pragma unroll
        for (int ks = 0; ks < 2; ks++) {
          int d0 = ks * 32 + boff;
          float4 f0 = *(const float4*)(kp + d0);
          float4 f1 = *(const float4*)(kp + d0 + 4);
          float4 f2 = *(const float4*)(kp + d0 + 8);
          float4 f3 = *(const float4*)(kp + d0 + 12);
          KB[ks].u[0] = pack_bf2(f0.x, f0.y); KB[ks].u[1] = pack_bf2(f0.z, f0.w);
          KB[ks].u[2] = pack_bf2(f1.x, f1.y); KB[ks].u[3] = pack_bf2(f1.z, f1.w);
          KB[ks].u[4] = pack_bf2(f2.x, f2.y); KB[ks].u[5] = pack_bf2(f2.z, f2.w);
          KB[ks].u[6] = pack_bf2(f3.x, f3.y); KB[ks].u[7] = pack_bf2(f3.z, f3.w);
        }
      }
      FragC ac;
#pragma unroll
      for (int v = 0; v < 8; v++) ac.f[v] = 0.f;
      ac.v = wmma_bf16(Qu[0].v, KB[0].v, ac.v);
      ac.v = wmma_bf16(Qu[1].v, KB[1].v, ac.v);

      // ---- rel-pos window: E[ii,c] = qv[ii] . r[p0-16+c], c in [0,32) ----
      const int p0 = (SEQ - 1) - I0 + Jt;
      FragC e01[2];
#pragma unroll
      for (int et = 0; et < 2; et++) {
        int p = p0 - 16 + et * 16 + lr;
        p = p < 0 ? 0 : (p > NPOS - 1 ? NPOS - 1 : p);
        const unsigned short* rp = Rh + (size_t)p * DHD;
        FragB RB[2];
#pragma unroll
        for (int ks = 0; ks < 2; ks++) {
          int d0 = ks * 32 + boff;
          uint4 x = *(const uint4*)(rp + d0);      // bf16 already
          uint4 y = *(const uint4*)(rp + d0 + 8);
          RB[ks].u[0] = x.x; RB[ks].u[1] = x.y; RB[ks].u[2] = x.z; RB[ks].u[3] = x.w;
          RB[ks].u[4] = y.x; RB[ks].u[5] = y.y; RB[ks].u[6] = y.z; RB[ks].u[7] = y.w;
        }
#pragma unroll
        for (int v = 0; v < 8; v++) e01[et].f[v] = 0.f;
        e01[et].v = wmma_bf16(Qv[0].v, RB[0].v, e01[et].v);
        e01[et].v = wmma_bf16(Qv[1].v, RB[1].v, e01[et].v);
      }

      // ---- XL shift: bd[ii,jj] = E[ii, 16+jj-ii] (same VGPR, same half) ----
#pragma unroll
      for (int v = 0; v < 8; v++) {
        int ii = v + aoff;
        int sl = (lane & 16) | ((lr - ii) & 15);
        float b0 = __shfl(e01[0].f[v], sl, 32);
        float b1 = __shfl(e01[1].f[v], sl, 32);
        float bd = (lr >= ii) ? b1 : b0;
        float sc = (ac.f[v] + bd) * 0.125f;                 // dh^-0.5
        sc = (Jt + lr > I0 + ii) ? -1e30f : sc;             // causal (log tril)
        s[kt].f[v] = sc;
      }
    }

    // ---- online softmax (rows live in (VGPR v, lane-half)) ----
    float mnew[8], corr[8];
#pragma unroll
    for (int v = 0; v < 8; v++) {
      float t = fmaxf(s[0].f[v], s[1].f[v]);
      t = fmaxf(t, __shfl_xor(t, 1, 32));
      t = fmaxf(t, __shfl_xor(t, 2, 32));
      t = fmaxf(t, __shfl_xor(t, 4, 32));
      t = fmaxf(t, __shfl_xor(t, 8, 32));
      mnew[v] = fmaxf(mrow[v], t);
      corr[v] = __expf(mrow[v] - mnew[v]);
      mrow[v] = mnew[v];
    }
#pragma unroll
    for (int v = 0; v < 8; v++) {
      s[0].f[v] = __expf(s[0].f[v] - mnew[v]);
      s[1].f[v] = __expf(s[1].f[v] - mnew[v]);
      acc[0].f[v] *= corr[v]; acc[1].f[v] *= corr[v];
      acc[2].f[v] *= corr[v]; acc[3].f[v] *= corr[v];
      lsum.f[v] *= corr[v];
    }

    // ---- stage P through LDS: D-layout -> A-layout (wave-private) ----
    unsigned short* sp16 = (unsigned short*)sP[wave];
#pragma unroll
    for (int v = 0; v < 8; v++) {
      int row = v + aoff;
      sp16[row * 32 + lr]      = f2bf(s[0].f[v]);
      sp16[row * 32 + 16 + lr] = f2bf(s[1].f[v]);
    }
    wavebar();
    FragB PA;
    {
      const unsigned int* pr = sP[wave] + lr * 16;
#pragma unroll
      for (int d = 0; d < 4; d++) PA.u[d] = pr[lh * 8 + d];
#pragma unroll
      for (int d = 0; d < 4; d++) PA.u[4 + d] = pr[lh * 8 + 4 + d];
    }
    wavebar();

    // ---- P @ V (+ row-sum via ones column) ----
#pragma unroll
    for (int t = 0; t < 4; t++) {
      FragB VB;
      const float* vp = Vp + (size_t)Jb * DHD + t * 16 + lr;
#pragma unroll
      for (int d = 0; d < 8; d++) {
        int k0 = boff + 2 * d;
        VB.u[d] = pack_bf2(vp[k0 * DHD], vp[(k0 + 1) * DHD]);
      }
      acc[t].v = wmma_bf16(PA.v, VB.v, acc[t].v);
    }
    lsum.v = wmma_bf16(PA.v, ones.v, lsum.v);
  }

  // ---- normalize and store ----
#pragma unroll
  for (int v = 0; v < 8; v++) {
    float l = __shfl(lsum.f[v], lh * 16, 32);  // denom lives in column 0
    float inv = 1.0f / l;
#pragma unroll
    for (int t = 0; t < 4; t++) {
      int row = I0 + v + aoff;
      out[((size_t)head * SEQ + row) * DHD + t * 16 + lr] = acc[t].f[v] * inv;
    }
  }
}

// ---------------------------------------------------------------------------
extern "C" void kernel_launch(void* const* d_in, const int* in_sizes, int n_in,
                              void* d_out, int out_size, void* d_ws, size_t ws_size,
                              hipStream_t stream) {
  (void)in_sizes; (void)n_in; (void)out_size; (void)ws_size;
  const float* seqs   = (const float*)d_in[0];
  const float* keysp  = (const float*)d_in[1];
  const float* vals   = (const float*)d_in[2];
  const float* ubias  = (const float*)d_in[3];
  const float* vbias  = (const float*)d_in[4];
  const float* rprojw = (const float*)d_in[5];
  // d_in[6] (attn_mask == log(tril)) is applied analytically in-kernel.
  float* out = (float*)d_out;

  float* enc = (float*)d_ws;                                       // 4.19 MB f32
  unsigned short* Rb =
      (unsigned short*)((char*)d_ws + (size_t)NPOS * DMODEL * 4);  // 2.10 MB bf16

  enc_kernel<<<(NPOS * DMODEL + 255) / 256, 256, 0, stream>>>(enc);
  rproj_kernel<<<512, 256, 0, stream>>>(enc, rprojw, Rb);          // 4096 tiles
  attn_kernel<<<128, 256, 0, stream>>>(seqs, keysp, vals, ubias, vbias, Rb, out);
}
